// KATBlocks_20014547599462
// MI455X (gfx1250) — matrix operand order repaired and checked by reading, not compiled, not run on hardware
//
#include <hip/hip_runtime.h>
#include <hip/hip_bf16.h>

// ---------------------------------------------------------------------------
// Model constants (fixed by the reference)
// ---------------------------------------------------------------------------
#define CB    2          // batch
#define CNX   4096       // patch tokens
#define CNK   256        // kernel tokens
#define CDIM  512
#define CH    8          // heads
#define CDH   64
#define CMLP  2048
#define CQKV  1536
#define CNPK  144.0f

#define USE_ASYNC_COPY 1

typedef __attribute__((ext_vector_type(16))) _Float16 v16h;
typedef __attribute__((ext_vector_type(8)))  _Float16 v8h;
typedef __attribute__((ext_vector_type(8)))  float    v8f;

// ---------------------------------------------------------------------------
// Device helpers
// ---------------------------------------------------------------------------
__device__ __forceinline__ float gelu_exact(float x) {
    return 0.5f * x * (1.0f + erff(x * 0.70710678118654752f));
}

__device__ __forceinline__ float blk_sum256(float v, float* red, int tid) {
    __syncthreads();
    red[tid] = v; __syncthreads();
    for (int s = 128; s > 0; s >>= 1) {
        if (tid < s) red[tid] += red[tid + s];
        __syncthreads();
    }
    float r = red[0]; __syncthreads();
    return r;
}

__device__ __forceinline__ float blk_max256(float v, float* red, int tid) {
    __syncthreads();
    red[tid] = v; __syncthreads();
    for (int s = 128; s > 0; s >>= 1) {
        if (tid < s) red[tid] = fmaxf(red[tid], red[tid + s]);
        __syncthreads();
    }
    float r = red[0]; __syncthreads();
    return r;
}

// 16-byte global->LDS copy, async (TENSOR/ASYNC path) or fallback.
__device__ __forceinline__ void copy16_g2l(const _Float16* g, _Float16* l) {
#if USE_ASYNC_COPY
    asm volatile("global_load_async_to_lds_b128 %0, %1, off"
                 :: "v"((unsigned)(unsigned long)l), "v"(g) : "memory");
#else
    *(v8h*)l = *(const v8h*)g;
#endif
}
__device__ __forceinline__ void async_wait0() {
#if USE_ASYNC_COPY
    asm volatile("s_wait_asynccnt 0x0" ::: "memory");
#endif
}

// ---------------------------------------------------------------------------
// Generic batched strided GEMM with WMMA (f16 inputs in memory, f32 accum)
//   C[z] = epi(alpha * A[z] @ B[z] + bias), optional f16 copy C16.
// EPI: 0 = +bias, 1 = gelu(+bias), 2 = res + gamma*(..), 3 = res + (..)
// TRANSB: B given as [N,K] row-major (q @ k^T without transposing k)
// WIDE:  64(M) x 128(N) tile, waves 2x4;  !WIDE: 128(M) x 64(N), waves 4x2.
// 256 threads = 8 wave32; wave tile 32x32 = 4 x v_wmma_f32_16x16x32_f16.
// Interior: double-buffered LDS pipeline, async global->LDS staging.
// NOTE: K is a multiple of 32 for all GEMMs here.
// ---------------------------------------------------------------------------
#define TK 32

template<int EPI, bool TRANSB, bool WIDE>
__global__ void __launch_bounds__(256)
gemm_wmma_kernel(const _Float16* __restrict__ A, long sAb, long sAh, int lda,
                 const _Float16* __restrict__ Bw, long sBb, long sBh, int ldb,
                 const float* __restrict__ bias,
                 const float* __restrict__ res, long sRb, long sRh, int ldres,
                 const float* __restrict__ gamma,
                 float* __restrict__ C, _Float16* __restrict__ C16,
                 long sCb, long sCh, int ldc,
                 int M, int N, int K, float alpha, int nh)
{
    constexpr int BM = WIDE ? 64 : 128;
    constexpr int BN = WIDE ? 128 : 64;
    constexpr int TPR8 = BN / 8;        // threads per B row (normal, v8h units)
    constexpr int RSTEP = 256 / TPR8;   // B row step per pass (normal)

    __shared__ __align__(16) _Float16 sA[2][BM][TK + 8];
    __shared__ __align__(16) _Float16 sB[2][TK][BN + 8];

    const int z  = blockIdx.z;
    const int bz = z / nh, hz = z % nh;
    const _Float16* Ab = A  + (long)bz * sAb + (long)hz * sAh;
    const _Float16* Bb = Bw + (long)bz * sBb + (long)hz * sBh;
    const float* Rb = res ? (res + (long)bz * sRb + (long)hz * sRh) : nullptr;

    const int m0   = blockIdx.y * BM;
    const int n0   = blockIdx.x * BN;
    const int tid  = threadIdx.x;
    const int lane = tid & 31;
    const int wave = tid >> 5;
    const int wm   = WIDE ? (wave >> 2) : (wave >> 1);
    const int wn   = WIDE ? (wave & 3)  : (wave & 1);

    const bool interior = (m0 + BM <= M) && (n0 + BN <= N);

    v8f acc[2][2] = {};

    const int arow  = lane & 15;
    const int khalf = (lane >> 4) << 3;   // 0 or 8 (A-fragment K sub-block)

    // staging coordinates (v8h = 16B granules)
    const int ar8  = tid >> 2;            // 0..63  A row base
    const int ac8  = (tid & 3) << 3;      // 0..24  A col
    const int br8  = tid / TPR8;          // B row base (normal)
    const int bc8  = (tid % TPR8) << 3;   // B col (normal)
    const int tn8  = tid >> 2;            // 0..63  B n-row base (TRANSB)
    const int tk8  = (tid & 3) << 3;      // 0..24  B k (TRANSB)

    auto stage = [&](int kk, int bufi) {
        #pragma unroll
        for (int p = 0; p < BM / 64; ++p)
            copy16_g2l(Ab + (long)(m0 + ar8 + 64 * p) * lda + kk + ac8,
                       &sA[bufi][ar8 + 64 * p][ac8]);
        if (TRANSB) {
            #pragma unroll
            for (int p = 0; p < BN / 64; ++p) {
                v8h h = *(const v8h*)(Bb + (long)(n0 + tn8 + 64 * p) * ldb + kk + tk8);
                #pragma unroll
                for (int q = 0; q < 8; ++q)
                    sB[bufi][tk8 + q][tn8 + 64 * p] = h[q];
            }
        } else {
            #pragma unroll
            for (int p = 0; p < BN / 64; ++p)
                copy16_g2l(Bb + (long)(kk + br8 + RSTEP * p) * ldb + n0 + bc8,
                           &sB[bufi][br8 + RSTEP * p][bc8]);
        }
    };

    if (interior) {
        // ---- double-buffered async pipeline ----
        stage(0, 0);
        async_wait0();
        __syncthreads();
        for (int k0 = 0; k0 < K; k0 += TK) {
            const int cur = (k0 / TK) & 1;
            const bool more = (k0 + TK) < K;
            if (more) stage(k0 + TK, 1 - cur);   // copies in flight during math

            v16h af[2], bf[2];
            #pragma unroll
            for (int fm = 0; fm < 2; ++fm) {
                int r = wm * 32 + fm * 16 + arow;
                #pragma unroll
                for (int e = 0; e < 16; ++e) {
                    int kk = ((e >> 3) << 4) + khalf + (e & 7);
                    af[fm][e] = sA[cur][r][kk];
                }
            }
            #pragma unroll
            for (int fn = 0; fn < 2; ++fn) {
                int nb = wn * 32 + fn * 16;
                #pragma unroll
                for (int e = 0; e < 16; ++e)
                    bf[fn][e] = sB[cur][((e >> 3) << 4) + khalf + (e & 7)][nb + arow];
            }
            #pragma unroll
            for (int fm = 0; fm < 2; ++fm)
                #pragma unroll
                for (int fn = 0; fn < 2; ++fn)
                    acc[fm][fn] = __builtin_amdgcn_wmma_f32_16x16x32_f16(
                        false, af[fm], false, bf[fn], (short)0, acc[fm][fn],
                        false, false);
            if (more) async_wait0();
            __syncthreads();
        }
    } else {
        // ---- guarded path (tiny boundary tiles: M=2 class-token GEMMs) ----
        for (int k0 = 0; k0 < K; k0 += TK) {
            for (int idx = tid; idx < BM * TK; idx += 256) {
                int r = idx >> 5, c = idx & 31;
                int gm = m0 + r;
                sA[0][r][c] = (gm < M) ? Ab[(long)gm * lda + k0 + c] : (_Float16)0.0f;
            }
            for (int idx = tid; idx < TK * BN; idx += 256) {
                int r = idx / BN, c = idx % BN;    // r = k, c = n
                int gn = n0 + c;
                _Float16 v;
                if (TRANSB) v = (gn < N) ? Bb[(long)gn * ldb + k0 + r] : (_Float16)0.0f;
                else        v = (gn < N) ? Bb[(long)(k0 + r) * ldb + gn] : (_Float16)0.0f;
                sB[0][r][c] = v;
            }
            __syncthreads();

            v16h af[2], bf[2];
            #pragma unroll
            for (int fm = 0; fm < 2; ++fm) {
                int r = wm * 32 + fm * 16 + arow;
                #pragma unroll
                for (int e = 0; e < 16; ++e) {
                    int kk = ((e >> 3) << 4) + khalf + (e & 7);
                    af[fm][e] = sA[0][r][kk];
                }
            }
            #pragma unroll
            for (int fn = 0; fn < 2; ++fn) {
                int nb = wn * 32 + fn * 16;
                #pragma unroll
                for (int e = 0; e < 16; ++e)
                    bf[fn][e] = sB[0][((e >> 3) << 4) + khalf + (e & 7)][nb + arow];
            }
            #pragma unroll
            for (int fm = 0; fm < 2; ++fm)
                #pragma unroll
                for (int fn = 0; fn < 2; ++fn)
                    acc[fm][fn] = __builtin_amdgcn_wmma_f32_16x16x32_f16(
                        false, af[fm], false, bf[fn], (short)0, acc[fm][fn],
                        false, false);
            __syncthreads();
        }
    }

    // ---- epilogue: D layout lane&15 = N, M = vgpr + 8*(lane>=16) ----
    float*     Cb   = C   ? (C   + (long)bz * sCb + (long)hz * sCh) : nullptr;
    _Float16*  Cb16 = C16 ? (C16 + (long)bz * sCb + (long)hz * sCh) : nullptr;
    const int nlo = lane & 15;
    const int mhi = (lane >> 4) << 3;
    #pragma unroll
    for (int fm = 0; fm < 2; ++fm)
        #pragma unroll
        for (int fn = 0; fn < 2; ++fn)
            #pragma unroll
            for (int e = 0; e < 8; ++e) {
                int m = m0 + wm * 32 + fm * 16 + mhi + e;
                int n = n0 + wn * 32 + fn * 16 + nlo;
                if (interior || (m < M && n < N)) {
                    float v = acc[fm][fn][e] * alpha;
                    if (bias) v += bias[n];
                    if (EPI == 1) v = gelu_exact(v);
                    if (EPI == 2) v = Rb[(long)m * ldres + n] + gamma[n] * v;
                    if (EPI == 3) v = Rb[(long)m * ldres + n] + v;
                    if (Cb)   Cb[(long)m * ldc + n]   = v;
                    if (Cb16) Cb16[(long)m * ldc + n] = (_Float16)v;
                }
            }
}

// ---------------------------------------------------------------------------
// fp32 -> f16 conversion (weights)
// ---------------------------------------------------------------------------
__global__ void __launch_bounds__(256)
cvt16_kernel(const float* __restrict__ in, _Float16* __restrict__ out, long n)
{
    long idx = (long)blockIdx.x * 256 + threadIdx.x;
    if (idx < n) out[idx] = (_Float16)in[idx];
}

// ---------------------------------------------------------------------------
// Depthwise 7-tap conv along token axis (zero padded) + bias
// ---------------------------------------------------------------------------
__global__ void __launch_bounds__(256)
dwconv_kernel(const float* __restrict__ x, const float* __restrict__ w,
              const float* __restrict__ b, float* __restrict__ out)
{
    long idx = (long)blockIdx.x * 256 + threadIdx.x;
    const long total = (long)CB * CNX * CDIM;
    if (idx >= total) return;
    int c  = (int)(idx % CDIM);
    int n  = (int)((idx / CDIM) % CNX);
    long bb = idx / ((long)CDIM * CNX);
    float acc = b[c];
    #pragma unroll
    for (int t = 0; t < 7; ++t) {
        int nn = n + t - 3;
        if (nn >= 0 && nn < CNX)
            acc += x[(bb * CNX + nn) * CDIM + c] * w[c * 7 + t];
    }
    out[idx] = acc;
}

// ---------------------------------------------------------------------------
// LayerNorm over last dim (=512); writes optional f32 and/or f16 outputs.
// ---------------------------------------------------------------------------
__global__ void __launch_bounds__(256)
ln512_kernel(const float* __restrict__ in, float* __restrict__ out,
             _Float16* __restrict__ out16,
             const float* __restrict__ g, const float* __restrict__ b,
             float eps)
{
    __shared__ float red[256];
    const int tid = threadIdx.x;
    const long row = blockIdx.x;
    const float* p = in + row * CDIM;
    float a0 = p[tid], a1 = p[tid + 256];
    float mean = blk_sum256(a0 + a1, red, tid) * (1.0f / CDIM);
    float d0 = a0 - mean, d1 = a1 - mean;
    float var = blk_sum256(d0 * d0 + d1 * d1, red, tid) * (1.0f / CDIM);
    float rs = rsqrtf(var + eps);
    float o0 = d0 * rs * g[tid]       + b[tid];
    float o1 = d1 * rs * g[tid + 256] + b[tid + 256];
    if (out)   { out[row * CDIM + tid] = o0;  out[row * CDIM + tid + 256] = o1; }
    if (out16) { out16[row * CDIM + tid] = (_Float16)o0;
                 out16[row * CDIM + tid + 256] = (_Float16)o1; }
}

// ---------------------------------------------------------------------------
// Softmax over NK=256 (patch->kernel) with mask + RBF decay; f16 output.
// grid: (NX, B*H), block 256
// ---------------------------------------------------------------------------
__global__ void __launch_bounds__(256)
softmax_pk_kernel(const float* __restrict__ s, _Float16* __restrict__ o16,
                  const float* __restrict__ mask, const float* __restrict__ kmask,
                  const float* __restrict__ rd, float dinv)
{
    __shared__ float red[256];
    const int i = blockIdx.x;
    const int z = blockIdx.y;
    const int b = z >> 3;
    const int j = threadIdx.x;
    const long base = ((long)z * CNX + i) * CNK;
    float v = s[base + j];
    if (mask[(long)b * CNX + i] * kmask[b * CNK + j] < 0.5f) v = -1e9f;
    float mx = blk_max256(v, red, j);
    float e  = expf(v - mx);
    float sm = blk_sum256(e, red, j);
    float p  = e / sm;
    float d  = rd[((long)b * CNK + j) * CNX + i];
    o16[base + j] = (_Float16)(p * expf(-d * d * dinv));
}

// ---------------------------------------------------------------------------
// Softmax over NX=4096 (kernel->patch) with mask + RBF decay; f16 output.
// grid: (NK, B*H), block 256, 16 elements per thread
// ---------------------------------------------------------------------------
__global__ void __launch_bounds__(256)
softmax_kp_kernel(const float* __restrict__ s, _Float16* __restrict__ o16,
                  const float* __restrict__ mask, const float* __restrict__ kmask,
                  const float* __restrict__ rd, float dinv)
{
    __shared__ float red[256];
    const int j = blockIdx.x;
    const int z = blockIdx.y;
    const int b = z >> 3;
    const int tid = threadIdx.x;
    const long base = ((long)z * CNK + j) * CNX;
    const float km = kmask[b * CNK + j];

    float lmax = -1e30f;
    for (int t = tid; t < CNX; t += 256) {
        float v = s[base + t];
        if (km * mask[(long)b * CNX + t] < 0.5f) v = -1e9f;
        lmax = fmaxf(lmax, v);
    }
    float mx = blk_max256(lmax, red, tid);

    float lsum = 0.0f;
    for (int t = tid; t < CNX; t += 256) {
        float v = s[base + t];
        if (km * mask[(long)b * CNX + t] < 0.5f) v = -1e9f;
        lsum += expf(v - mx);
    }
    float sm = blk_sum256(lsum, red, tid);

    for (int t = tid; t < CNX; t += 256) {
        float v = s[base + t];
        if (km * mask[(long)b * CNX + t] < 0.5f) v = -1e9f;
        float p = expf(v - mx) / sm;
        float d = rd[((long)b * CNK + j) * CNX + t];
        o16[base + t] = (_Float16)(p * expf(-d * d * dinv));
    }
}

// ---------------------------------------------------------------------------
// Class-token attention (1 query per batch); f16 in/out, f32 math.
// grid: B blocks, 256 threads; loops over heads.
// ---------------------------------------------------------------------------
__global__ void __launch_bounds__(256)
class_attn_kernel(const _Float16* __restrict__ qkvc, const _Float16* __restrict__ qkvk,
                  const float* __restrict__ mask, const float* __restrict__ kmask,
                  _Float16* __restrict__ cmerged)
{
    __shared__ float red[256];
    __shared__ float sp[256];
    const int b = blockIdx.x;
    const int tid = threadIdx.x;
    const int j = tid;
    for (int h = 0; h < CH; ++h) {
        const _Float16* cq  = qkvc + b * CQKV + h * CDH;
        const _Float16* kkp = qkvk + ((long)(b * CNK + j)) * CQKV + CDIM + h * CDH;
        float dot = 0.0f;
        for (int d = 0; d < CDH; ++d) dot += (float)cq[d] * (float)kkp[d];
        dot *= 0.125f;   // 1/sqrt(64)
        if (mask[(long)b * CNX] * kmask[b * CNK + j] < 0.5f) dot = -1e9f;
        float mx = blk_max256(dot, red, tid);
        float e  = expf(dot - mx);
        float sm = blk_sum256(e, red, tid);
        sp[j] = e / sm;
        __syncthreads();
        if (tid < CDH) {
            float o = 0.0f;
            for (int j2 = 0; j2 < CNK; ++j2)
                o += sp[j2] * (float)qkvk[((long)(b * CNK + j2)) * CQKV + 2 * CDIM + h * CDH + tid];
            cmerged[b * CDIM + h * CDH + tid] = (_Float16)o;
        }
        __syncthreads();
    }
}

// ---------------------------------------------------------------------------
// k_reps write: out = where(kmask < 0.5, 0, kx)
// ---------------------------------------------------------------------------
__global__ void __launch_bounds__(256)
kreps_kernel(const float* __restrict__ kx, const float* __restrict__ kmask,
             float* __restrict__ out)
{
    long idx = (long)blockIdx.x * 256 + threadIdx.x;
    const long total = (long)CB * CNK * CDIM;
    if (idx >= total) return;
    int j  = (int)((idx / CDIM) % CNK);
    long b = idx / ((long)CDIM * CNK);
    out[idx] = (kmask[b * CNK + j] < 0.5f) ? 0.0f : kx[idx];
}

// ---------------------------------------------------------------------------
// Host-side GEMM dispatch
// ---------------------------------------------------------------------------
struct GP {
    const _Float16* A;  long sAb, sAh; int lda;
    const _Float16* Bw; long sBb, sBh; int ldb;
    const float* bias;
    const float* res; long sRb, sRh; int ldres;
    const float* gamma;
    float* C; _Float16* C16; long sCb, sCh; int ldc;
    int M, N, K; float alpha; int nh; int nb;
};

template<int EPI, bool TB, bool WIDE>
static void run_gemm(const GP& g, hipStream_t st) {
    const int bn = WIDE ? 128 : 64;
    const int bm = WIDE ? 64 : 128;
    dim3 grid((g.N + bn - 1) / bn, (g.M + bm - 1) / bm, g.nb);
    gemm_wmma_kernel<EPI, TB, WIDE><<<grid, dim3(256), 0, st>>>(
        g.A, g.sAb, g.sAh, g.lda, g.Bw, g.sBb, g.sBh, g.ldb, g.bias,
        g.res, g.sRb, g.sRh, g.ldres, g.gamma,
        g.C, g.C16, g.sCb, g.sCh, g.ldc, g.M, g.N, g.K, g.alpha, g.nh);
}

static void gemm_plain(int epi, const _Float16* A, const _Float16* W, const float* bias,
                       const float* res, const float* gamma,
                       float* C, _Float16* C16,
                       int M, int N, int K, hipStream_t st) {
    GP g{A, 0, 0, K, W, 0, 0, N, bias, res, 0, 0, N, gamma,
         C, C16, 0, 0, N, M, N, K, 1.0f, 1, 1};
    switch (epi) {
        case 0: run_gemm<0, false, true>(g, st); break;
        case 1: run_gemm<1, false, true>(g, st); break;
        case 2: run_gemm<2, false, true>(g, st); break;
        default: run_gemm<3, false, true>(g, st); break;
    }
}

static void cvt16(const float* in, _Float16* out, long n, hipStream_t st) {
    cvt16_kernel<<<dim3((unsigned)((n + 255) / 256)), 256, 0, st>>>(in, out, n);
}

// ---------------------------------------------------------------------------
// kernel_launch
// ---------------------------------------------------------------------------
extern "C" void kernel_launch(void* const* d_in, const int* in_sizes, int n_in,
                              void* d_out, int out_size, void* d_ws, size_t ws_size,
                              hipStream_t stream) {
    (void)in_sizes; (void)n_in; (void)out_size; (void)ws_size;

    const float* in_x     = (const float*)d_in[0];
    const float* in_kx    = (const float*)d_in[1];
    const float* in_rd    = (const float*)d_in[2];
    const float* in_clst  = (const float*)d_in[3];
    const float* in_mask  = (const float*)d_in[4];
    const float* in_kmask = (const float*)d_in[5];
    const float* dw_w  = (const float*)d_in[6];
    const float* dw_b  = (const float*)d_in[7];
    const float* cn_g  = (const float*)d_in[8];
    const float* cn_b  = (const float*)d_in[9];
    const float* pw1_w = (const float*)d_in[10];
    const float* pw1_b = (const float*)d_in[11];
    const float* pw2_w = (const float*)d_in[12];
    const float* pw2_b = (const float*)d_in[13];
    const float* gammaw= (const float*)d_in[14];
    const float* pn_g  = (const float*)d_in[15];
    const float* pn_b  = (const float*)d_in[16];
    const float* qkv_w = (const float*)d_in[17];
    const float* out_w = (const float*)d_in[18];
    const float* out_b = (const float*)d_in[19];
    const float* ff_g  = (const float*)d_in[20];
    const float* ff_b  = (const float*)d_in[21];
    const float* ff1_w = (const float*)d_in[22];
    const float* ff1_b = (const float*)d_in[23];
    const float* ff2_w = (const float*)d_in[24];
    const float* ff2_b = (const float*)d_in[25];

    float* outp = (float*)d_out;

    // ---- workspace carving ----
    const long NROWS = (long)CB * CNX;                    // 8192
    const long nBND  = NROWS * CDIM;                      // 4.19M
    const long nKR   = (long)CB * CNK * CDIM;             // 262144
    const long nSC   = (long)CB * CH * CNX * CNK;         // 16.78M
    const long nQX   = NROWS * (long)CQKV;
    const long nQK   = (long)CB * CNK * CQKV;

    char* cur = (char*)d_ws;
    auto alloc = [&](long bytes) {
        char* p = cur;
        cur += (bytes + 255) & ~255L;
        return (void*)p;
    };
    float* xbuf    = (float*)alloc(nBND * 4);
    float* kxbuf   = (float*)alloc(nKR * 4);
    float* clstbuf = (float*)alloc(CB * CDIM * 4);
    float* region1 = (float*)alloc(nSC * 4);   // sc1 (f32) / t2048h (f16)
    float* region2 = (float*)alloc(nSC * 4);   // sc2 (f32) / hbuf (f32)
    float* sc1 = region1;  _Float16* t2048h = (_Float16*)region1;
    float* sc2 = region2;  float* hbuf = region2;
    _Float16* sc1h  = (_Float16*)alloc(nSC * 2);
    _Float16* sc2h  = (_Float16*)alloc(nSC * 2);  _Float16* hbuf16 = sc2h; // disjoint in time
    _Float16* xbuf16   = (_Float16*)alloc(nBND * 2);
    _Float16* kx16     = (_Float16*)alloc(nKR * 2);
    _Float16* clst16   = (_Float16*)alloc(CB * CDIM * 2);
    _Float16* xtmp16   = (_Float16*)alloc(nBND * 2);
    _Float16* kxtmp16  = (_Float16*)alloc(nKR * 2);
    _Float16* clsttmp16= (_Float16*)alloc(CB * CDIM * 2);
    _Float16* qkvx16   = (_Float16*)alloc(nQX * 2);
    _Float16* qkvk16   = (_Float16*)alloc(nQK * 2);
    _Float16* qkvc16   = (_Float16*)alloc(CB * CQKV * 2);
    _Float16* amerged16= (_Float16*)alloc(nBND * 2);
    _Float16* kmerged16= (_Float16*)alloc(nKR * 2);
    _Float16* cmerged16= (_Float16*)alloc(CB * CDIM * 2);
    _Float16* pw1w16 = (_Float16*)alloc(2L * CDIM * CMLP * 2);
    _Float16* pw2w16 = (_Float16*)alloc(2L * CMLP * CDIM * 2);
    _Float16* qkvw16 = (_Float16*)alloc(2L * CDIM * CQKV * 2);
    _Float16* outw16 = (_Float16*)alloc(2L * CDIM * CDIM * 2);
    _Float16* ff1w16 = (_Float16*)alloc(2L * CDIM * CMLP * 2);
    _Float16* ff2w16 = (_Float16*)alloc(2L * CMLP * CDIM * 2);

    // ---- weight conversion (f32 -> f16), both layers ----
    cvt16(pw1_w, pw1w16, 2L * CDIM * CMLP, stream);
    cvt16(pw2_w, pw2w16, 2L * CMLP * CDIM, stream);
    cvt16(qkv_w, qkvw16, 2L * CDIM * CQKV, stream);
    cvt16(out_w, outw16, 2L * CDIM * CDIM, stream);
    cvt16(ff1_w, ff1w16, 2L * CDIM * CMLP, stream);
    cvt16(ff2_w, ff2w16, 2L * CMLP * CDIM, stream);

    // ---- init working copies of mutable tensors ----
    hipMemcpyAsync(xbuf,    in_x,    nBND * 4,      hipMemcpyDeviceToDevice, stream);
    hipMemcpyAsync(kxbuf,   in_kx,   nKR * 4,       hipMemcpyDeviceToDevice, stream);
    hipMemcpyAsync(clstbuf, in_clst, CB * CDIM * 4, hipMemcpyDeviceToDevice, stream);

    for (int l = 0; l < 2; ++l) {
        const float* dw_w_l  = dw_w  + (long)l * CDIM * 7;
        const float* dw_b_l  = dw_b  + l * CDIM;
        const float* cn_g_l  = cn_g  + l * CDIM;
        const float* cn_b_l  = cn_b  + l * CDIM;
        const _Float16* pw1w_l = pw1w16 + (long)l * CDIM * CMLP;
        const float* pw1_b_l = pw1_b + l * CMLP;
        const _Float16* pw2w_l = pw2w16 + (long)l * CMLP * CDIM;
        const float* pw2_b_l = pw2_b + l * CDIM;
        const float* gamma_l = gammaw+ l * CDIM;
        const float* pn_g_l  = pn_g  + l * CDIM;
        const float* pn_b_l  = pn_b  + l * CDIM;
        const _Float16* qkvw_l = qkvw16 + (long)l * CDIM * CQKV;
        const _Float16* outw_l = outw16 + (long)l * CDIM * CDIM;
        const float* out_b_l = out_b + l * CDIM;
        const float* ff_g_l  = ff_g  + l * CDIM;
        const float* ff_b_l  = ff_b  + l * CDIM;
        const _Float16* ff1w_l = ff1w16 + (long)l * CDIM * CMLP;
        const float* ff1_b_l = ff1_b + l * CMLP;
        const _Float16* ff2w_l = ff2w16 + (long)l * CMLP * CDIM;
        const float* ff2_b_l = ff2_b + l * CDIM;

        const float dinv = 1.0f / (2.0f * CNPK * (float)(1 << l));

        // ---- ConvNeXt block: x = x + gamma * pw2(gelu(pw1(ln(dwconv(x))))) ----
        dwconv_kernel<<<dim3((unsigned)((nBND + 255) / 256)), 256, 0, stream>>>(xbuf, dw_w_l, dw_b_l, hbuf);
        ln512_kernel<<<dim3((unsigned)NROWS), 256, 0, stream>>>(hbuf, nullptr, hbuf16, cn_g_l, cn_b_l, 1e-6f);
        gemm_plain(1, hbuf16, pw1w_l, pw1_b_l, nullptr, nullptr, nullptr, t2048h, (int)NROWS, CMLP, CDIM, stream);
        gemm_plain(2, t2048h, pw2w_l, pw2_b_l, xbuf, gamma_l, xbuf, nullptr, (int)NROWS, CDIM, CMLP, stream);

        // ---- pre-attention LayerNorms (kept, per reference) ----
        ln512_kernel<<<dim3((unsigned)NROWS), 256, 0, stream>>>(xbuf, xbuf, xbuf16, pn_g_l, pn_b_l, 1e-5f);
        ln512_kernel<<<dim3(CB * CNK), 256, 0, stream>>>(kxbuf, kxbuf, kx16, pn_g_l, pn_b_l, 1e-5f);
        ln512_kernel<<<dim3(CB), 256, 0, stream>>>(clstbuf, clstbuf, clst16, pn_g_l, pn_b_l, 1e-5f);

        // ---- QKV projections (f16 outputs only) ----
        gemm_plain(0, xbuf16, qkvw_l, nullptr, nullptr, nullptr, nullptr, qkvx16, (int)NROWS, CQKV, CDIM, stream);
        gemm_plain(0, kx16,   qkvw_l, nullptr, nullptr, nullptr, nullptr, qkvk16, CB * CNK,   CQKV, CDIM, stream);
        gemm_plain(0, clst16, qkvw_l, nullptr, nullptr, nullptr, nullptr, qkvc16, CB,         CQKV, CDIM, stream);

        // ---- patch -> kernel attention ----
        {   // scores1[z,i,j] = 0.125 * tq . kk   (f32 out for softmax)
            GP g{qkvx16, (long)CNX * CQKV, CDH, CQKV,
                 qkvk16 + CDIM, (long)CNK * CQKV, CDH, CQKV,
                 nullptr, nullptr, 0, 0, 0, nullptr,
                 sc1, nullptr, (long)CH * CNX * CNK, (long)CNX * CNK, CNK,
                 CNX, CNK, CDH, 0.125f, CH, CB * CH};
            run_gemm<0, true, true>(g, stream);
        }
        softmax_pk_kernel<<<dim3(CNX, CB * CH), 256, 0, stream>>>(sc1, sc1h, in_mask, in_kmask, in_rd, dinv);
        {   // att_out merged (N=64 -> narrow tile)
            GP g{sc1h, (long)CH * CNX * CNK, (long)CNX * CNK, CNK,
                 qkvk16 + 2 * CDIM, (long)CNK * CQKV, CDH, CQKV,
                 nullptr, nullptr, 0, 0, 0, nullptr,
                 nullptr, amerged16, (long)CNX * CDIM, CDH, CDIM,
                 CNX, CDH, CNK, 1.0f, CH, CB * CH};
            run_gemm<0, false, false>(g, stream);
        }
        gemm_plain(3, amerged16, outw_l, out_b_l, xbuf, nullptr, xbuf, nullptr, (int)NROWS, CDIM, CDIM, stream);

        // ---- kernel -> patch attention ----
        {   // scores2[z,j,i] = 0.125 * kq . tk
            GP g{qkvk16, (long)CNK * CQKV, CDH, CQKV,
                 qkvx16 + CDIM, (long)CNX * CQKV, CDH, CQKV,
                 nullptr, nullptr, 0, 0, 0, nullptr,
                 sc2, nullptr, (long)CH * CNK * CNX, (long)CNK * CNX, CNX,
                 CNK, CNX, CDH, 0.125f, CH, CB * CH};
            run_gemm<0, true, true>(g, stream);
        }
        softmax_kp_kernel<<<dim3(CNK, CB * CH), 256, 0, stream>>>(sc2, sc2h, in_mask, in_kmask, in_rd, dinv);
        {   // k_out merged (N=64 -> narrow tile)
            GP g{sc2h, (long)CH * CNK * CNX, (long)CNK * CNX, CNX,
                 qkvx16 + 2 * CDIM, (long)CNX * CQKV, CDH, CQKV,
                 nullptr, nullptr, 0, 0, 0, nullptr,
                 nullptr, kmerged16, (long)CNK * CDIM, CDH, CDIM,
                 CNK, CDH, CNX, 1.0f, CH, CB * CH};
            run_gemm<0, false, false>(g, stream);
        }
        gemm_plain(3, kmerged16, outw_l, out_b_l, kxbuf, nullptr, kxbuf, nullptr, CB * CNK, CDIM, CDIM, stream);

        // ---- class -> kernel attention ----
        class_attn_kernel<<<dim3(CB), 256, 0, stream>>>(qkvc16, qkvk16, in_mask, in_kmask, cmerged16);
        gemm_plain(3, cmerged16, outw_l, out_b_l, clstbuf, nullptr, clstbuf, nullptr, CB, CDIM, CDIM, stream);

        // ---- FFN on x, kx, clst (t = t + pw2(gelu(pw1(ln(t))))) ----
        ln512_kernel<<<dim3((unsigned)NROWS), 256, 0, stream>>>(xbuf, nullptr, xtmp16, ff_g_l, ff_b_l, 1e-5f);
        gemm_plain(1, xtmp16, ff1w_l, ff1_b_l, nullptr, nullptr, nullptr, t2048h, (int)NROWS, CMLP, CDIM, stream);
        gemm_plain(3, t2048h, ff2w_l, ff2_b_l, xbuf, nullptr, xbuf, nullptr, (int)NROWS, CDIM, CMLP, stream);

        ln512_kernel<<<dim3(CB * CNK), 256, 0, stream>>>(kxbuf, nullptr, kxtmp16, ff_g_l, ff_b_l, 1e-5f);
        gemm_plain(1, kxtmp16, ff1w_l, ff1_b_l, nullptr, nullptr, nullptr, t2048h, CB * CNK, CMLP, CDIM, stream);
        gemm_plain(3, t2048h, ff2w_l, ff2_b_l, kxbuf, nullptr, kxbuf, nullptr, CB * CNK, CDIM, CMLP, stream);

        ln512_kernel<<<dim3(CB), 256, 0, stream>>>(clstbuf, nullptr, clsttmp16, ff_g_l, ff_b_l, 1e-5f);
        gemm_plain(1, clsttmp16, ff1w_l, ff1_b_l, nullptr, nullptr, nullptr, t2048h, CB, CMLP, CDIM, stream);
        gemm_plain(3, t2048h, ff2w_l, ff2_b_l, clstbuf, nullptr, clstbuf, nullptr, CB, CDIM, CMLP, stream);

        // ---- k_reps[l] = where(kmask < 0.5, 0, kx) ----
        kreps_kernel<<<dim3((unsigned)((nKR + 255) / 256)), 256, 0, stream>>>(
            kxbuf, in_kmask, outp + (long)l * nKR);
    }

    // ---- final clst output ----
    hipMemcpyAsync(outp + 2 * nKR, clstbuf, CB * CDIM * 4,
                   hipMemcpyDeviceToDevice, stream);
}
